// XNORConv2d_70334384439650
// MI455X (gfx1250) — 8-bit matrix operand order repaired and checked by reading, compiled, not run on hardware
//
#include <hip/hip_runtime.h>
#include <stdint.h>

typedef __attribute__((ext_vector_type(8))) int v8i;

#define BATCH 32
#define C_IN  256
#define H_IN  58
#define W_IN  58
#define O_OUT 256
#define H_OUT 56
#define W_OUT 56
#define KTOT  2304
#define M_TOT (BATCH * H_OUT * W_OUT)
#define BLK_M 64
#define BLK_N 256
#define K_STEP 64
#define LDS_PITCH 80

__device__ __forceinline__ void async_copy_b128(const void* lds_ptr, const void* gptr) {
    uint32_t lds_addr = (uint32_t)(uintptr_t)lds_ptr;
    uint64_t gaddr    = (uint64_t)(uintptr_t)gptr;
    asm volatile("global_load_async_to_lds_b128 %0, %1, off"
                 :: "v"(lds_addr), "v"(gaddr) : "memory");
}
#define WAIT_ASYNCCNT(n) asm volatile("s_wait_asynccnt %0" :: "n"(n) : "memory")

__global__ void pack_x_kernel(const float* __restrict__ x, uint8_t* __restrict__ xu8) {
    long idx = (long)blockIdx.x * blockDim.x + threadIdx.x;
    const long total = (long)BATCH * C_IN * H_IN * W_IN;
    if (idx >= total) return;
    int w = (int)(idx % W_IN); long t = idx / W_IN;
    int h = (int)(t % H_IN);   t /= H_IN;
    int c = (int)(t % C_IN);   int b = (int)(t / C_IN);
    uint8_t v = (x[idx] >= 0.5f) ? (uint8_t)1 : (uint8_t)0;
    xu8[(((long)b * H_IN + h) * W_IN + w) * C_IN + c] = v;
}

__global__ void pack_w_kernel(const float* __restrict__ wt, int8_t* __restrict__ wpk,
                              float* __restrict__ scale, float* __restrict__ beta) {
    const int o = blockIdx.x;
    const int c = threadIdx.x;
    float sabs = 0.0f;
    int   cnt  = 0;
    #pragma unroll
    for (int tap = 0; tap < 9; ++tap) {
        int kh = tap / 3, kw = tap % 3;
        float w = wt[(((long)o * C_IN + c) * 3 + kh) * 3 + kw];
        sabs += fabsf(w);
        int bit = (w >= 0.0f) ? 1 : 0;
        cnt += bit;
        wpk[(long)o * KTOT + tap * C_IN + c] = (int8_t)(2 * bit - 1);
    }
    __shared__ float sred[256];
    __shared__ int   cred[256];
    sred[c] = sabs; cred[c] = cnt;
    __syncthreads();
    for (int s = 128; s > 0; s >>= 1) {
        if (c < s) { sred[c] += sred[c + s]; cred[c] += cred[c + s]; }
        __syncthreads();
    }
    if (c == 0) {
        float alpha = sred[0] / (float)(C_IN * H_IN * W_IN);
        scale[o] = 2.0f * alpha;
        beta[o]  = alpha * (float)(KTOT - 2 * cred[0]);
    }
}

__global__ __launch_bounds__(256) void xnor_wmma_kernel(
        const uint8_t* __restrict__ xu8, const int8_t* __restrict__ wpk,
        const float* __restrict__ scale, const float* __restrict__ beta,
        float* __restrict__ out) {
    __shared__ __align__(16) uint8_t lA[2][BLK_M][LDS_PITCH];
    __shared__ __align__(16) uint8_t lB[2][BLK_N][LDS_PITCH];

    const int tid   = threadIdx.x;
    const int lane  = tid & 31;
    const int wv    = tid >> 5;
    const int waveM = wv & 1;
    const int waveN = wv >> 1;
    const int lm    = lane & 15;
    const int hi    = lane >> 4;

    const int mBase = blockIdx.x * BLK_M;

    const int aRow = tid >> 2;
    const int aOff = (tid & 3) * 16;
    const int mG   = mBase + aRow;
    const int bIdx = mG / (H_OUT * W_OUT);
    const int rem  = mG % (H_OUT * W_OUT);
    const int oy   = rem / W_OUT;
    const int ox   = rem % W_OUT;

    auto issue_stage = [&](int k0, int buf) {
        const int tap = k0 >> 8;
        const int kh = tap / 3, kw = tap % 3;
        const int c0 = (k0 & 255) + aOff;
        const uint8_t* ga = xu8 +
            ((((long)bIdx * H_IN + (oy + kh)) * W_IN + (ox + kw)) * C_IN + c0);
        async_copy_b128(&lA[buf][aRow][aOff], ga);
        const int8_t* gb = wpk + (long)tid * KTOT + k0;
        #pragma unroll
        for (int j = 0; j < 4; ++j)
            async_copy_b128(&lB[buf][tid][j * 16], gb + j * 16);
    };

    v8i acc[2][4];
    #pragma unroll
    for (int s = 0; s < 2; ++s)
        #pragma unroll
        for (int n = 0; n < 4; ++n)
            acc[s][n] = (v8i)0;

    issue_stage(0, 0);
    int cur = 0;
    for (int k0 = 0; k0 < KTOT; k0 += K_STEP) {
        const int nxt = cur ^ 1;
        const bool more = (k0 + K_STEP) < KTOT;
        if (more) {
            issue_stage(k0 + K_STEP, nxt);
            WAIT_ASYNCCNT(5);
        } else {
            WAIT_ASYNCCNT(0);
        }
        __syncthreads();

        v8i afr[2];
        #pragma unroll
        for (int s = 0; s < 2; ++s) {
            const uint32_t* rp = (const uint32_t*)(&lA[cur][waveM * 32 + s * 16 + lm][0]);
            #pragma unroll
            for (int i = 0; i < 8; ++i) {
                int off = ((i >> 1) * 16 + hi * 8 + (i & 1) * 4) >> 2;
                afr[s][i] = (int)rp[off];
            }
        }
        v8i bfr[4];
        #pragma unroll
        for (int n = 0; n < 4; ++n) {
            const uint32_t* cp = (const uint32_t*)(&lB[cur][waveN * 64 + n * 16 + lm][0]);
            #pragma unroll
            for (int i = 0; i < 8; ++i) {
                int off = ((i >> 1) * 16 + hi * 8 + (i & 1) * 4) >> 2;
                bfr[n][i] = (int)cp[off];
            }
        }

        #pragma unroll
        for (int s = 0; s < 2; ++s)
            #pragma unroll
            for (int n = 0; n < 4; ++n)
                acc[s][n] = __builtin_amdgcn_wmma_i32_16x16x64_iu8(
                     false, afr[s],  true, bfr[n],
                    acc[s][n],  false,  false);

        __syncthreads();
        cur = nxt;
    }

    #pragma unroll
    for (int n = 0; n < 4; ++n) {
        const int o  = waveN * 64 + n * 16 + lm;
        const float sc = scale[o];
        const float be = beta[o];
        #pragma unroll
        for (int s = 0; s < 2; ++s) {
            #pragma unroll
            for (int r = 0; r < 8; ++r) {
                int m  = mBase + waveM * 32 + s * 16 + r + 8 * hi;
                int bb = m / (H_OUT * W_OUT);
                int rr = m % (H_OUT * W_OUT);
                int yy = rr / W_OUT;
                int xx = rr % W_OUT;
                out[(((long)bb * O_OUT + o) * H_OUT + yy) * W_OUT + xx] =
                    sc * (float)acc[s][n][r] + be;
            }
        }
    }
}

extern "C" void kernel_launch(void* const* d_in, const int* in_sizes, int n_in,
                              void* d_out, int out_size, void* d_ws, size_t ws_size,
                              hipStream_t stream) {
    (void)in_sizes; (void)n_in; (void)out_size; (void)ws_size;
    const float* x  = (const float*)d_in[0];
    const float* wt = (const float*)d_in[1];
    float* out = (float*)d_out;

    const size_t XU8_BYTES = (size_t)BATCH * H_IN * W_IN * C_IN;
    const size_t WPK_BYTES = (size_t)O_OUT * KTOT;
    uint8_t* xu8  = (uint8_t*)d_ws;
    int8_t*  wpk  = (int8_t*)((char*)d_ws + XU8_BYTES);
    float*   sclp = (float*)((char*)d_ws + XU8_BYTES + WPK_BYTES);
    float*   betp = sclp + O_OUT;

    const long nx = (long)BATCH * C_IN * H_IN * W_IN;
    pack_x_kernel<<<dim3((unsigned)((nx + 255) / 256)), dim3(256), 0, stream>>>(x, xu8);
    pack_w_kernel<<<dim3(O_OUT), dim3(256), 0, stream>>>(wt, wpk, sclp, betp);
    xnor_wmma_kernel<<<dim3(M_TOT / BLK_M), dim3(256), 0, stream>>>(xu8, wpk, sclp, betp, out);
}
